// stk_pred_small_1_77910706749811
// MI455X (gfx1250) — compile-verified, matrix operand order checked by reading, not run on hardware
//
#include <hip/hip_runtime.h>
#include <hip/hip_bf16.h>

typedef __attribute__((ext_vector_type(16))) _Float16 v16h;
typedef __attribute__((ext_vector_type(8)))  _Float16 v8h;
typedef __attribute__((ext_vector_type(2)))  _Float16 v2h;
typedef __attribute__((ext_vector_type(8)))  float    v8f;

#define N_CTX 2048
#define DIM   128
#define STK   4000
#define TILE_J 32
#define NT_J  (N_CTX / TILE_J)
#define ROW_HALVES 40   // padded LDS row stride in halves (80 B): 16B aligned, conflict-free b128 reads

union BFrag { v16h v; v8h h[2]; };

// Stage x[jt:jt+32, 0:128] -> buf[d][k] transposed + converted to f16 (cooperative, 256 threads)
__device__ __forceinline__ void stage_tile(const float* __restrict__ x, size_t bN, int jt,
                                           int tid, _Float16* buf)
{
    const int d   = tid & 127;
    const int kp0 = tid >> 7;               // 0..1
#pragma unroll
    for (int it = 0; it < 8; ++it) {
        const int k0 = (kp0 + it * 2) * 2;  // even k, covers pair (k0, k0+1)
        const float a0 = x[(bN + jt + k0    ) * DIM + d];
        const float a1 = x[(bN + jt + k0 + 1) * DIM + d];
        v2h pk; pk[0] = (_Float16)a0; pk[1] = (_Float16)a1;
        *(v2h*)&buf[d * ROW_HALVES + k0] = pk;
    }
}

__global__ __launch_bounds__(256) void stk_flash_wmma_kernel(
    const float* __restrict__ x,           // (B, N, D) f32
    const int*   __restrict__ stk_ten,     // (B, N)    i32
    const float* __restrict__ stk_matrix,  // (S, S)    f32
    const float* __restrict__ stk_weight,  // (S, S)    f32
    float*       __restrict__ out)         // (B, N, D) f32
{
    __shared__ __align__(16) _Float16 xT[2][DIM * ROW_HALVES];   // double-buffered x tile

    const int b        = blockIdx.y;            // 0..7
    const int rowBlock = blockIdx.x;            // 0..15 (128 rows each)
    const int tid      = threadIdx.x;
    const int lane     = tid & 31;
    const int wave     = tid >> 5;              // 0..7
    const int g        = (lane >> 4) & 1;       // half-wave group
    const int lan15    = lane & 15;
    const int srcBase  = (lane & 16) >> 1;      // +8 row offset for upper half-wave (C layout)

    const size_t bN    = (size_t)b * N_CTX;
    const int    iBase = rowBlock * 128 + wave * 16;
    const int    ipos  = iBase + lan15;         // global row position this lane owns

    const int    rowIdx = stk_ten[bN + ipos];
    const size_t rowOff = (size_t)rowIdx * STK;

    v8f acc[8];
#pragma unroll
    for (int nt = 0; nt < 8; ++nt) acc[nt] = (v8f){0.f,0.f,0.f,0.f,0.f,0.f,0.f,0.f};

    float runmax = 0.0f;   // W = |.| >= 0 so 0 is a valid floor (diag entry is exactly 0)
    float runsum = 0.0f;

    stage_tile(x, bN, 0, tid, xT[0]);
    __syncthreads();

    for (int t = 0; t < NT_J; ++t) {
        const int jt = t * TILE_J;
        const _Float16* cur = xT[t & 1];

        // Prefetch next x tile into the other buffer (overlaps with gather/softmax/WMMA below)
        if (t + 1 < NT_J) stage_tile(x, bN, jt + TILE_J, tid, xT[(t + 1) & 1]);

        // ---- column indices for this j-tile, broadcast via bpermute ----
        const int cidx = stk_ten[bN + jt + lane];

        // ---- gather W (abs) and M in WMMA A-fragment element order ----
        float wv[16], mv[16];
#pragma unroll
        for (int e = 0; e < 16; ++e) {
            const int k   = ((e < 8) ? e : e + 8) + 8 * g;   // A-fragment K for element e
            const int col = __shfl(cidx, k, 32);
            const size_t off = rowOff + (size_t)col;
            wv[e] = __builtin_fabsf(stk_weight[off]);
            mv[e] = stk_matrix[off];
        }

        // ---- positional diagonal: intersects exactly one j-tile per wave (uniform skip) ----
        if (jt < iBase + 16 && jt + TILE_J > iBase) {
#pragma unroll
            for (int e = 0; e < 16; ++e) {
                const int k = ((e < 8) ? e : e + 8) + 8 * g;
                if (jt + k == ipos) { wv[e] = 0.0f; mv[e] = 0.0f; }
            }
        }

        // ---- row max of this tile ----
        float tmax = wv[0];
#pragma unroll
        for (int e = 1; e < 16; ++e) tmax = fmaxf(tmax, wv[e]);
        tmax = fmaxf(tmax, __shfl_xor(tmax, 16, 32));

        // ---- rescale only when some row's max actually grew (wave-uniform branch) ----
        if (__any(tmax > runmax)) {
            const float nmax  = fmaxf(runmax, tmax);
            const float scale = __expf(runmax - nmax);
            runmax  = nmax;
            runsum *= scale;
            float sv[8];
#pragma unroll
            for (int v = 0; v < 8; ++v) sv[v] = __shfl(scale, v + srcBase, 32);
#pragma unroll
            for (int nt = 0; nt < 8; ++nt)
#pragma unroll
                for (int v = 0; v < 8; ++v) acc[nt][v] *= sv[v];
        }

        // ---- probabilities, A fragment = m * exp(w - runmax) in f16 ----
        float psum = 0.0f;
        v16h afrag;
#pragma unroll
        for (int e = 0; e < 16; ++e) {
            const float p = __expf(wv[e] - runmax);
            psum += p;
            afrag[e] = (_Float16)(mv[e] * p);
        }
        psum  += __shfl_xor(psum, 16, 32);
        runsum += psum;

        // ---- 8 WMMAs: (16x32 A) x (32x16 B per 16-wide slice of D) ----
#pragma unroll
        for (int nt = 0; nt < 8; ++nt) {
            const _Float16* bp = &cur[(nt * 16 + lan15) * ROW_HALVES];
            BFrag bu;
            bu.h[0] = *(const v8h*)(bp + 8 * g);             // K = 8g .. 8g+7
            bu.h[1] = *(const v8h*)(bp + 16 + 8 * g);        // K = 16+8g .. 23+8g
            acc[nt] = __builtin_amdgcn_wmma_f32_16x16x32_f16(
                false, afrag, false, bu.v, (short)0, acc[nt], false, false);
        }

        __syncthreads();   // next tile's staging done + this tile's compute done
    }

    // ---- finalize: divide by softmax denominator, store ----
    const float inv = 1.0f / runsum;
    float iv[8];
#pragma unroll
    for (int v = 0; v < 8; ++v) iv[v] = __shfl(inv, v + srcBase, 32);

#pragma unroll
    for (int nt = 0; nt < 8; ++nt) {
        const int d = nt * 16 + lan15;
#pragma unroll
        for (int v = 0; v < 8; ++v) {
            const int row = v + srcBase;                     // C layout: vgpr v, halves split
            out[(bN + iBase + row) * DIM + d] = acc[nt][v] * iv[v];
        }
    }
}

extern "C" void kernel_launch(void* const* d_in, const int* in_sizes, int n_in,
                              void* d_out, int out_size, void* d_ws, size_t ws_size,
                              hipStream_t stream) {
    const float* x          = (const float*)d_in[0];
    const int*   stk_ten    = (const int*)  d_in[1];
    const float* stk_matrix = (const float*)d_in[2];
    const float* stk_weight = (const float*)d_in[3];
    float* out = (float*)d_out;

    dim3 grid(16, 8);   // 16 row-blocks of 128 rows x 8 batches
    dim3 block(256);    // 8 waves, each owning a 16-row i-tile
    stk_flash_wmma_kernel<<<grid, block, 0, stream>>>(x, stk_ten, stk_matrix, stk_weight, out);
}
